// RFCNRoIhead_50663434224549
// MI455X (gfx1250) — matrix operand order repaired and checked by reading, not compiled, not run on hardware
//
#include <hip/hip_runtime.h>
#include <hip/hip_bf16.h>

// ---------------------------------------------------------------------------
// R-FCN RoI head, restructured:
//   h1 = h @ w1^T (+b1)          -> big bf16 WMMA GEMM (5056 x 1024 x 1024),
//                                   double-buffered global_load_async_to_lds
//   integral image of h1         -> O(1) box sums
//   PS-RoI pooled mid features P -> [300][49*1024 + 49 mask + pad] bf16
//   out = (1/49) * P @ W2^T      -> small bf16 WMMA GEMM (300 x 32 x 50240)
// ---------------------------------------------------------------------------

typedef __attribute__((ext_vector_type(16))) __bf16 v16bf;
typedef __attribute__((ext_vector_type(8)))  __bf16 bf16x8;
typedef __attribute__((ext_vector_type(8)))  float  v8f;

#define CIN   1024
#define CMID  1024
#define NS    5000      // N*H*W = 2*50*50
#define NSP   5056      // padded to 79*64: no bounds checks in gemm1
#define HH    50
#define WW    50
#define RR    300
#define KP    50240     // 49*1024 + 49 (mask) + 15 pad -> multiple of 32
#define KFEAT 50176     // 49*1024
#define KMASK 50225     // 49*1024 + 49

static __device__ __forceinline__ v16bf cat8(bf16x8 lo, bf16x8 hi) {
    return __builtin_shufflevector(lo, hi, 0,1,2,3,4,5,6,7,8,9,10,11,12,13,14,15);
}

// ---- h [n][c][yx] f32  ->  hb [n*2500+yx][c] bf16 (LDS-tiled transpose) ---
__global__ __launch_bounds__(256) void convert_h_kernel(const float* __restrict__ h,
                                                        __bf16* __restrict__ hb) {
    __shared__ float tile[32][33];
    const int n   = blockIdx.z;
    const int c0  = blockIdx.x * 32;
    const int yx0 = blockIdx.y * 32;
    const int t   = threadIdx.x;
#pragma unroll
    for (int p = 0; p < 4; ++p) {
        int idx = t + p * 256;
        int i = idx >> 5;                  // c local
        int j = idx & 31;                  // yx local
        int yx = yx0 + j;
        tile[i][j] = (yx < 2500) ? h[(size_t)(n * CIN + c0 + i) * 2500 + yx] : 0.0f;
    }
    __syncthreads();
#pragma unroll
    for (int p = 0; p < 4; ++p) {
        int idx = t + p * 256;
        int j = idx >> 5;                  // yx local
        int i = idx & 31;                  // c local
        int yx = yx0 + j;
        if (yx < 2500)
            hb[(size_t)(n * 2500 + yx) * CIN + c0 + i] = (__bf16)tile[i][j];
    }
}

// ---- convert w1 [o][k] f32 -> bf16 ----------------------------------------
__global__ __launch_bounds__(256) void convert_w1_kernel(const float* __restrict__ w,
                                                         __bf16* __restrict__ wb) {
    int idx = blockIdx.x * 256 + threadIdx.x;
    if (idx >= CMID * CIN) return;
    wb[idx] = (__bf16)w[idx];
}

// ---- big GEMM: h1[s][o] = sum_k hb[s][k] * w1b[o][k] + b1[o] --------------
// M = s (64), N = o (64) per 256-thread block (8 waves, 2 subtiles each).
// Double-buffered LDS; tiles staged with GLOBAL_LOAD_ASYNC_TO_LDS_B128 and
// pipelined against WMMA via s_wait_asynccnt thresholds.
__global__ __launch_bounds__(256) void gemm1_kernel(const __bf16* __restrict__ hb,   // [NSP][CIN]
                                                    const __bf16* __restrict__ w1b,  // [CMID][CIN]
                                                    const float*  __restrict__ b1,
                                                    float* __restrict__ h1) {        // [NSP][CMID]
    __shared__ __bf16 As0[64][40];  // [row][k_local], 80B row stride (bank skew)
    __shared__ __bf16 As1[64][40];
    __shared__ __bf16 Bs0[64][40];
    __shared__ __bf16 Bs1[64][40];

    const int s0 = blockIdx.x * 64;
    const int o0 = blockIdx.y * 64;
    const int t      = threadIdx.x;
    const int wave   = t >> 5;
    const int lane   = t & 31;
    const int lane16 = lane & 15;
    const int hi     = lane >> 4;
    const int m_sub  = wave & 3;     // s subtile 0..3
    const int n_pair = wave >> 2;    // o subtile pair 0..1

    // staging assignment: thread copies one 16B chunk per tile per K-step
    const int row = t >> 2;          // 0..63
    const int kq  = (t & 3) * 8;     // 0,8,16,24 (halves)

    const unsigned ldsA0 = (unsigned)(uintptr_t)&As0[row][kq];
    const unsigned ldsA1 = (unsigned)(uintptr_t)&As1[row][kq];
    const unsigned ldsB0 = (unsigned)(uintptr_t)&Bs0[row][kq];
    const unsigned ldsB1 = (unsigned)(uintptr_t)&Bs1[row][kq];
    const __bf16* ga = hb  + (size_t)(s0 + row) * CIN + kq;
    const __bf16* gb = w1b + (size_t)(o0 + row) * CIN + kq;

    v8f acc0 = {};
    v8f acc1 = {};

    const int am = m_sub * 16 + lane16;
    const int bn0 = n_pair * 32 + lane16;
    const int bn1 = bn0 + 16;

    auto issue = [&](unsigned la, unsigned lb, int k) {
        asm volatile("global_load_async_to_lds_b128 %0, %1, off"
                     :: "v"(la), "v"(ga + k) : "memory");
        asm volatile("global_load_async_to_lds_b128 %0, %1, off"
                     :: "v"(lb), "v"(gb + k) : "memory");
    };
    auto mm_step = [&](const __bf16 (*Asb)[40], const __bf16 (*Bsb)[40]) {
        v16bf af = cat8(*(const bf16x8*)(&Asb[am][hi * 8]),
                        *(const bf16x8*)(&Asb[am][16 + hi * 8]));
        v16bf bf0 = cat8(*(const bf16x8*)(&Bsb[bn0][hi * 8]),
                         *(const bf16x8*)(&Bsb[bn0][16 + hi * 8]));
        v16bf bf1 = cat8(*(const bf16x8*)(&Bsb[bn1][hi * 8]),
                         *(const bf16x8*)(&Bsb[bn1][16 + hi * 8]));
        acc0 = __builtin_amdgcn_wmma_f32_16x16x32_bf16(false, af, false, bf0,
                                                       (short)0, acc0, false, false);
        acc1 = __builtin_amdgcn_wmma_f32_16x16x32_bf16(false, af, false, bf1,
                                                       (short)0, acc1, false, false);
    };

    // prologue: buffer 0 <- K chunk 0
    issue(ldsA0, ldsB0, 0);

    for (int k0 = 0; k0 < CIN; k0 += 64) {
        // buffer 1 <- K chunk k0+32 (always valid: CIN is a multiple of 64)
        issue(ldsA1, ldsB1, k0 + 32);
        // wait for buffer-0 copies only; buffer-1 DMA overlaps the WMMAs below
        asm volatile("s_wait_asynccnt 0x2" ::: "memory");
        __syncthreads();
        mm_step(As0, Bs0);
        __syncthreads();                 // all waves done reading buffer 0

        if (k0 + 64 < CIN) {
            issue(ldsA0, ldsB0, k0 + 64);    // buffer 0 <- K chunk k0+64
            asm volatile("s_wait_asynccnt 0x2" ::: "memory");
        } else {
            asm volatile("s_wait_asynccnt 0x0" ::: "memory");
        }
        __syncthreads();
        mm_step(As1, Bs1);
        __syncthreads();                 // all waves done reading buffer 1
    }

    // epilogue: D layout -> VGPR r, lane L: M(s) = r + 8*(L/16), N(o) = L%16
#pragma unroll
    for (int j = 0; j < 2; ++j) {
        const int o = o0 + (n_pair * 2 + j) * 16 + lane16;
        const v8f& acc = (j == 0) ? acc0 : acc1;
        const float bias = b1[o];
#pragma unroll
        for (int r = 0; r < 8; ++r) {
            const int s = s0 + m_sub * 16 + hi * 8 + r;
            h1[(size_t)s * CMID + o] = acc[r] + bias;
        }
    }
}

// ---- integral image: I[n][y][x][c] = sum_{yy<y, xx<x} h1 ------------------
__global__ __launch_bounds__(256) void integral_kernel(const float* __restrict__ h1,
                                                       float* __restrict__ I) {
    int idx = blockIdx.x * 256 + threadIdx.x;   // 0..2047
    int n = idx >> 10;
    int c = idx & 1023;
    const size_t nb = (size_t)n * 51 * 51 * 1024;
    for (int x = 0; x <= WW; ++x) I[nb + (size_t)x * 1024 + c] = 0.0f;
    for (int y = 1; y <= HH; ++y) I[nb + (size_t)(y * 51) * 1024 + c] = 0.0f;
    for (int y = 0; y < HH; ++y) {
        float rs = 0.0f;
        for (int x = 0; x < WW; ++x) {
            rs += h1[(size_t)(n * 2500 + y * WW + x) * 1024 + c];
            I[nb + (size_t)((y + 1) * 51 + (x + 1)) * 1024 + c] =
                I[nb + (size_t)(y * 51 + (x + 1)) * 1024 + c] + rs;
        }
    }
}

// ---- PS-RoI pool of h1 via integral image -> P (bf16) ---------------------
__global__ __launch_bounds__(256) void pool_kernel(const float* __restrict__ I,
                                                   const float* __restrict__ rois,
                                                   const int*   __restrict__ roi_idx,
                                                   __bf16* __restrict__ P) {
    const int r   = blockIdx.y;
    const int bin = blockIdx.x;        // ph*7+pw
    const int ph  = bin / 7, pw = bin - ph * 7;

    const float x1 = rois[r * 4 + 0], y1 = rois[r * 4 + 1];
    const float x2 = rois[r * 4 + 2], y2 = rois[r * 4 + 3];
    const float sw = rintf(x1) * 0.0625f;
    const float sh = rintf(y1) * 0.0625f;
    const float ew = rintf(x2 + 1.0f) * 0.0625f;
    const float eh = rintf(y2 + 1.0f) * 0.0625f;
    const float bw = fmaxf(ew - sw, 0.1f) * (1.0f / 7.0f);
    const float bh = fmaxf(eh - sh, 0.1f) * (1.0f / 7.0f);

    const int hs = (int)fminf(fmaxf(floorf((float)ph * bh + sh), 0.0f), 50.0f);
    const int he = (int)fminf(fmaxf(ceilf(((float)ph + 1.0f) * bh + sh), 0.0f), 50.0f);
    const int ws = (int)fminf(fmaxf(floorf((float)pw * bw + sw), 0.0f), 50.0f);
    const int we = (int)fminf(fmaxf(ceilf(((float)pw + 1.0f) * bw + sw), 0.0f), 50.0f);
    const int cnt = (he - hs) * (we - ws);
    const float inv = (cnt > 0) ? 1.0f / (float)cnt : 0.0f;

    const int n = roi_idx[r];
    const size_t nb = (size_t)n * 51 * 51 * 1024;
    const float* i11 = I + nb + (size_t)(he * 51 + we) * 1024;
    const float* i01 = I + nb + (size_t)(hs * 51 + we) * 1024;
    const float* i10 = I + nb + (size_t)(he * 51 + ws) * 1024;
    const float* i00 = I + nb + (size_t)(hs * 51 + ws) * 1024;

    __bf16* prow = P + (size_t)r * KP + bin * 1024;
    for (int c = threadIdx.x; c < 1024; c += 256) {
        float s = i11[c] - i01[c] - i10[c] + i00[c];
        prow[c] = (__bf16)(s * inv);
    }
    if (threadIdx.x == 0)
        P[(size_t)r * KP + KFEAT + bin] = (__bf16)((cnt > 0) ? 1.0f : 0.0f);
    if (bin == 0 && threadIdx.x < (KP - KMASK))
        P[(size_t)r * KP + KMASK + threadIdx.x] = (__bf16)0.0f;
}

// ---- build W2 [32][KP] bf16 from wc/bc/wl/bl ------------------------------
__global__ __launch_bounds__(256) void w2_kernel(const float* __restrict__ wc,
                                                 const float* __restrict__ bc,
                                                 const float* __restrict__ wl,
                                                 const float* __restrict__ bl,
                                                 __bf16* __restrict__ W2) {
    int idx = blockIdx.x * 256 + threadIdx.x;
    if (idx >= 32 * KP) return;
    int o  = idx / KP;
    int kk = idx - o * KP;
    float v = 0.0f;
    if (o < 25) {
        if (kk < KFEAT) {
            int bin = kk >> 10, c = kk & 1023;
            v = (o < 21) ? wc[(size_t)(o * 49 + bin) * 1024 + c]
                         : wl[(size_t)((o - 21) * 49 + bin) * 1024 + c];
        } else if (kk < KMASK) {
            int bin = kk - KFEAT;
            v = (o < 21) ? bc[o * 49 + bin] : bl[(o - 21) * 49 + bin];
        }
    }
    W2[idx] = (__bf16)v;
}

// ---- final GEMM: out = (1/49) * P @ W2^T, M=300 N=32 K=KP -----------------
// one wave per block; 16 rows x 32 cols per block; fragments straight from L2.
__global__ __launch_bounds__(32) void gemm2_kernel(const __bf16* __restrict__ P,
                                                   const __bf16* __restrict__ W2,
                                                   float* __restrict__ out) {
    const int m0 = blockIdx.x * 16;
    const int lane = threadIdx.x;
    const int lane16 = lane & 15;
    const int hi = lane >> 4;

    const int  m  = m0 + lane16;
    const bool mv = (m < RR);
    const __bf16* prow = P  + (size_t)m * KP;
    const __bf16* w0   = W2 + (size_t)lane16 * KP;
    const __bf16* w1r  = W2 + (size_t)(16 + lane16) * KP;

    v8f acc0 = {};
    v8f acc1 = {};
    for (int k0 = 0; k0 < KP; k0 += 32) {
        __builtin_prefetch(prow + k0 + 256, 0, 1);
        __builtin_prefetch(w0 + k0 + 256, 0, 1);
        v16bf af = {};
        if (mv)
            af = cat8(*(const bf16x8*)(prow + k0 + hi * 8),
                      *(const bf16x8*)(prow + k0 + 16 + hi * 8));
        v16bf bf0 = cat8(*(const bf16x8*)(w0 + k0 + hi * 8),
                         *(const bf16x8*)(w0 + k0 + 16 + hi * 8));
        v16bf bf1 = cat8(*(const bf16x8*)(w1r + k0 + hi * 8),
                         *(const bf16x8*)(w1r + k0 + 16 + hi * 8));
        acc0 = __builtin_amdgcn_wmma_f32_16x16x32_bf16(false, af, false, bf0,
                                                       (short)0, acc0, false, false);
        acc1 = __builtin_amdgcn_wmma_f32_16x16x32_bf16(false, af, false, bf1,
                                                       (short)0, acc1, false, false);
    }

    const float sc = 1.0f / 49.0f;
#pragma unroll
    for (int rI = 0; rI < 8; ++rI) {
        int mm = m0 + hi * 8 + rI;
        if (mm < RR) {
            int   n0 = lane16;             // 0..15  -> classes 0..15
            out[mm * 21 + n0] = acc0[rI] * sc;
            int   n1 = 16 + lane16;        // 16..31 -> classes 16..20, regs 21..24
            float v1 = acc1[rI] * sc;
            if (n1 < 21)       out[mm * 21 + n1] = v1;
            else if (n1 < 25)  out[6300 + mm * 4 + (n1 - 21)] = v1;
        }
    }
}

// ---------------------------------------------------------------------------
extern "C" void kernel_launch(void* const* d_in, const int* in_sizes, int n_in,
                              void* d_out, int out_size, void* d_ws, size_t ws_size,
                              hipStream_t stream) {
    (void)in_sizes; (void)n_in; (void)out_size; (void)ws_size;
    const float* h    = (const float*)d_in[0];
    const float* rois = (const float*)d_in[1];
    const int*   ridx = (const int*)  d_in[2];
    const float* w1   = (const float*)d_in[3];
    const float* b1   = (const float*)d_in[4];
    const float* wc   = (const float*)d_in[5];
    const float* bc   = (const float*)d_in[6];
    const float* wl   = (const float*)d_in[7];
    const float* bl   = (const float*)d_in[8];
    float* out = (float*)d_out;

    char* ws = (char*)d_ws;
    size_t off = 0;
    auto carve = [&](size_t bytes) -> void* {
        void* p = ws + off;
        off += (bytes + 255) & ~(size_t)255;
        return p;
    };
    __bf16* hb  = (__bf16*)carve((size_t)NSP * CIN * 2);           // 10.35 MB
    __bf16* w1b = (__bf16*)carve((size_t)CMID * CIN * 2);          //  2.10 MB
    float*  h1  = (float*) carve((size_t)NSP * CMID * 4);          // 20.71 MB
    float*  I   = (float*) carve((size_t)2 * 51 * 51 * 1024 * 4);  // 21.31 MB
    __bf16* P   = (__bf16*)carve((size_t)RR * KP * 2);             // 30.14 MB
    __bf16* W2  = (__bf16*)carve((size_t)32 * KP * 2);             //  3.22 MB

    convert_h_kernel <<<dim3(CIN / 32, 79, 2), 256, 0, stream>>>(h, hb);
    convert_w1_kernel<<<(CMID * CIN + 255) / 256, 256, 0, stream>>>(w1, w1b);
    gemm1_kernel     <<<dim3(NSP / 64, CMID / 64), 256, 0, stream>>>(hb, w1b, b1, h1);
    integral_kernel  <<<8, 256, 0, stream>>>(h1, I);
    pool_kernel      <<<dim3(49, RR), 256, 0, stream>>>(I, rois, ridx, P);
    w2_kernel        <<<(32 * KP + 255) / 256, 256, 0, stream>>>(wc, bc, wl, bl, W2);
    gemm2_kernel     <<<(RR + 15) / 16, 32, 0, stream>>>(P, W2, out);
}
